// NeRFNetwork_57200374448766
// MI455X (gfx1250) — matrix operand order repaired and checked by reading, not compiled, not run on hardware
//
#include <hip/hip_runtime.h>

typedef _Float16 half_t;
typedef half_t v16h __attribute__((ext_vector_type(16)));
typedef half_t v8h  __attribute__((ext_vector_type(8)));
typedef float  v8f  __attribute__((ext_vector_type(8)));

#define NPTS (1 << 20)
#define TBL  (1 << 19)
#define TMASK (TBL - 1)
#define LVLS 16
#define PTS_PER_BLOCK 128
#define THREADS 128

// padded f16 weight offsets (in halves) inside workspace / LDS
#define OFF_WS0 0        // sigma_w0  [32][32]  (in 19 -> pad 32)
#define OFF_WS1 1024     // sigma_w1  [16][32]  (row 0 valid)
#define OFF_WC0 1536     // color_w0  [64][64]  (in 35 -> pad 64)
#define OFF_WC1 5632     // color_w1  [64][64]
#define OFF_WC2 9728     // color_w2  [16][64]  (rows 0-5 valid)
#define OFF_WP0 10752    // spec_w0   [32][32]  (in 6 -> pad 32)
#define OFF_WP1 11776    // spec_w1   [16][32]  (rows 0-2 valid)
#define W_TOTAL 12288    // halves

// ---------------- weight prep: f32 -> padded f16 in workspace ----------------
__global__ void prep_weights(const float* __restrict__ sw0, const float* __restrict__ sw1,
                             const float* __restrict__ cw0, const float* __restrict__ cw1,
                             const float* __restrict__ cw2, const float* __restrict__ pw0,
                             const float* __restrict__ pw1, half_t* __restrict__ W)
{
    int tid = threadIdx.x; // single block of 256
    for (int i = tid; i < 1024; i += 256) { int r = i >> 5, c = i & 31; W[OFF_WS0 + i] = (half_t)((c < 19) ? sw0[r * 19 + c] : 0.f); }
    for (int i = tid; i < 512;  i += 256) { int r = i >> 5, c = i & 31; W[OFF_WS1 + i] = (half_t)((r == 0) ? sw1[c] : 0.f); }
    for (int i = tid; i < 4096; i += 256) { int r = i >> 6, c = i & 63; W[OFF_WC0 + i] = (half_t)((c < 35) ? cw0[r * 35 + c] : 0.f); }
    for (int i = tid; i < 4096; i += 256) { int r = i >> 6, c = i & 63; W[OFF_WC1 + i] = (half_t)(cw1[r * 64 + c]); }
    for (int i = tid; i < 1024; i += 256) { int r = i >> 6, c = i & 63; W[OFF_WC2 + i] = (half_t)((r < 6) ? cw2[r * 64 + c] : 0.f); }
    for (int i = tid; i < 1024; i += 256) { int r = i >> 5, c = i & 31; W[OFF_WP0 + i] = (half_t)((c < 6) ? pw0[r * 6 + c] : 0.f); }
    for (int i = tid; i < 512;  i += 256) { int r = i >> 5, c = i & 31; W[OFF_WP1 + i] = (half_t)((r < 3) ? pw1[r * 32 + c] : 0.f); }
}

// ---------------- WMMA fragment helpers (CDNA5 16x16x32 f16 layouts) ----------
__device__ __forceinline__ void lds_fence() { asm volatile("s_wait_dscnt 0" ::: "memory"); }

__device__ __forceinline__ v16h cat8(v8h lo, v8h hi) {
    v16h r;
#pragma unroll
    for (int i = 0; i < 8; ++i) { r[i] = lo[i]; r[i + 8] = hi[i]; }
    return r;
}

// A 16x32 f16: lane<16 holds row=lane K{0..7,16..23}; lane>=16 row=lane-16 K{8..15,24..31}
__device__ __forceinline__ v16h load_a16(const half_t* base, int strideH, int kc) {
    int lane = threadIdx.x & 31, row = lane & 15, sel = lane >> 4;
    const half_t* p = base + row * strideH + kc * 32 + sel * 8;
    v8h lo = *(const v8h*)p;
    v8h hi = *(const v8h*)(p + 16);
    return cat8(lo, hi);
}

// B 32x16 f16 from row-major weights w[out][Kp]: lane<16 col=lane K{0..7,16..23}, lane>=16 K{8..15,24..31}
__device__ __forceinline__ v16h load_b16(const half_t* w, int Kp, int n0, int kc) {
    int lane = threadIdx.x & 31, col = lane & 15, sel = lane >> 4;
    const half_t* p = w + (n0 + col) * Kp + kc * 32 + sel * 8;
    v8h lo = *(const v8h*)p;
    v8h hi = *(const v8h*)(p + 16);
    return cat8(lo, hi);
}

// D (f32, lane n holds col n for rows r and r+8) -> relu -> f16 row-major LDS
__device__ __forceinline__ void store_h(half_t* h, int strideH, int n0, v8f acc) {
    int lane = threadIdx.x & 31, col = (lane & 15) + n0, sel = lane >> 4;
#pragma unroll
    for (int r = 0; r < 8; ++r) {
        float v = fmaxf(acc[r], 0.f);
        h[(r + sel * 8) * strideH + col] = (half_t)v;
    }
}

#define WMMA(A, B, C) __builtin_amdgcn_wmma_f32_16x16x32_f16(false, (A), false, (B), (short)0, (C), false, false)

__device__ __forceinline__ float sigmoidf(float x) { return 1.f / (1.f + expf(-x)); }

// ---------------- fused NeRF kernel ------------------------------------------
__global__ void __launch_bounds__(THREADS)
nerf_fused(const float* __restrict__ xg, const float* __restrict__ dg,
           const float* __restrict__ embD, const float* __restrict__ embC,
           const half_t* __restrict__ Wg, float* __restrict__ out)
{
    __shared__ __align__(16) half_t Wl[W_TOTAL];                 // 24 KB weights
    __shared__ __align__(16) half_t Asig[PTS_PER_BLOCK][32];     // 8 KB sigma inputs
    __shared__ __align__(16) half_t Acol[PTS_PER_BLOCK][64];     // 16 KB color inputs
    __shared__ __align__(16) half_t Hid[4][16 * 64];             // 8 KB per-wave hidden scratch

    const int tid = threadIdx.x;

    // stage weights into LDS (16 B per thread iteration)
    {
        const uint4* s = (const uint4*)Wg;
        uint4* q = (uint4*)Wl;
        for (int i = tid; i < W_TOTAL / 8; i += THREADS) q[i] = s[i];
    }

    // ---------- phase 1: hash-grid encoding, one point per thread ----------
    const int pt = blockIdx.x * PTS_PER_BLOCK + tid;
    const float x0 = xg[pt * 3 + 0], x1 = xg[pt * 3 + 1], x2 = xg[pt * 3 + 2];
    Asig[tid][0] = (half_t)x0; Asig[tid][1] = (half_t)x1; Asig[tid][2] = (half_t)x2;
    Acol[tid][0] = (half_t)x0; Acol[tid][1] = (half_t)x1; Acol[tid][2] = (half_t)x2;
#pragma unroll
    for (int c = 19; c < 32; ++c) Asig[tid][c] = (half_t)0.f;
#pragma unroll
    for (int c = 35; c < 64; ++c) Acol[tid][c] = (half_t)0.f;

    const float u0 = (x0 + 1.f) * 0.5f, u1 = (x1 + 1.f) * 0.5f, u2 = (x2 + 1.f) * 0.5f;
    const unsigned PRIME1 = 2654435761u, PRIME2 = 805459861u;
    float resf = 16.0f;
#pragma unroll 1
    for (int lvl = 0; lvl < LVLS; ++lvl) {
        float res = floorf(resf);
        resf *= 1.38191288f;   // exp(ln(2048/16)/15)
        float px = u0 * res, py = u1 * res, pz = u2 * res;
        float fx = floorf(px), fy = floorf(py), fz = floorf(pz);
        float wx = px - fx, wy = py - fy, wz = pz - fz;
        unsigned ux = (unsigned)fx, uy = (unsigned)fy, uz = (unsigned)fz;
        unsigned hx0 = ux, hx1 = ux + 1u;
        unsigned hy0 = uy * PRIME1, hy1 = hy0 + PRIME1;
        unsigned hz0 = uz * PRIME2, hz1 = hz0 + PRIME2;
        const float* tD = embD + (size_t)lvl * TBL;
        const float* tC = embC + (size_t)lvl * TBL * 2;
        float aD = 0.f, aC0 = 0.f, aC1 = 0.f;
#pragma unroll
        for (int c = 0; c < 8; ++c) {
            unsigned h = (((c & 4) ? hx1 : hx0) ^ ((c & 2) ? hy1 : hy0) ^ ((c & 1) ? hz1 : hz0)) & TMASK;
            float wgt = ((c & 4) ? wx : 1.f - wx) * ((c & 2) ? wy : 1.f - wy) * ((c & 1) ? wz : 1.f - wz);
            aD += tD[h] * wgt;
            float2 f2 = *(const float2*)(tC + 2u * h);
            aC0 += f2.x * wgt; aC1 += f2.y * wgt;
        }
        Asig[tid][3 + lvl]         = (half_t)aD;
        Acol[tid][3 + 2 * lvl]     = (half_t)aC0;
        Acol[tid][3 + 2 * lvl + 1] = (half_t)aC1;
    }

    __syncthreads();

    // ---------- phase 2: WMMA MLPs, one 16-point row-tile at a time ----------
    const int wave = tid >> 5, lane = tid & 31;
    const int col = lane & 15, sel = lane >> 4;
    half_t* hid = &Hid[wave][0];
    const half_t* WS0 = Wl + OFF_WS0;
    const half_t* WS1 = Wl + OFF_WS1;
    const half_t* WC0 = Wl + OFF_WC0;
    const half_t* WC1 = Wl + OFF_WC1;
    const half_t* WC2 = Wl + OFF_WC2;
    const half_t* WP0 = Wl + OFF_WP0;
    const half_t* WP1 = Wl + OFF_WP1;

#pragma unroll 1
    for (int t = 0; t < 2; ++t) {
        const int m0 = wave * 32 + t * 16;
        const int pt0 = blockIdx.x * PTS_PER_BLOCK + m0;

        // ---- sigma MLP: [19->32] relu [32->1] exp ----
        v8f s0 = {}, s1 = {};
        {
            v16h a = load_a16(&Asig[m0][0], 32, 0);
            s0 = WMMA(a, load_b16(WS0, 32, 0, 0), s0);
            s1 = WMMA(a, load_b16(WS0, 32, 16, 0), s1);
        }
        store_h(hid, 64, 0, s0);
        store_h(hid, 64, 16, s1);
        lds_fence();
        v8f sg = {};
        {
            v16h a = load_a16(hid, 64, 0);
            sg = WMMA(a, load_b16(WS1, 32, 0, 0), sg);
        }
        if (col == 0) {
#pragma unroll
            for (int r = 0; r < 8; ++r) out[pt0 + r + sel * 8] = expf(sg[r]);
        }

        // ---- color MLP layer0: [35->64] relu ----
        v8f c0 = {}, c1 = {}, c2 = {}, c3 = {};
#pragma unroll
        for (int kc = 0; kc < 2; ++kc) {
            v16h a = load_a16(&Acol[m0][0], 64, kc);
            c0 = WMMA(a, load_b16(WC0, 64, 0, kc), c0);
            c1 = WMMA(a, load_b16(WC0, 64, 16, kc), c1);
            c2 = WMMA(a, load_b16(WC0, 64, 32, kc), c2);
            c3 = WMMA(a, load_b16(WC0, 64, 48, kc), c3);
        }
        store_h(hid, 64, 0, c0); store_h(hid, 64, 16, c1);
        store_h(hid, 64, 32, c2); store_h(hid, 64, 48, c3);
        lds_fence();

        // ---- color layer1: [64->64] relu ----
        v8f d0 = {}, d1 = {}, d2 = {}, d3 = {};
#pragma unroll
        for (int kc = 0; kc < 2; ++kc) {
            v16h a = load_a16(hid, 64, kc);
            d0 = WMMA(a, load_b16(WC1, 64, 0, kc), d0);
            d1 = WMMA(a, load_b16(WC1, 64, 16, kc), d1);
            d2 = WMMA(a, load_b16(WC1, 64, 32, kc), d2);
            d3 = WMMA(a, load_b16(WC1, 64, 48, kc), d3);
        }
        store_h(hid, 64, 0, d0); store_h(hid, 64, 16, d1);
        store_h(hid, 64, 32, d2); store_h(hid, 64, 48, d3);
        lds_fence();

        // ---- color layer2: [64->6] logits kept in regs ----
        v8f e = {};
#pragma unroll
        for (int kc = 0; kc < 2; ++kc) {
            v16h a = load_a16(hid, 64, kc);
            e = WMMA(a, load_b16(WC2, 64, 0, kc), e);
        }
        lds_fence();

        // ---- build specular input [d(3), sigmoid(spec_feat)(3), 0..] in hid ----
#pragma unroll
        for (int r = 0; r < 8; ++r) {
            int row = r + sel * 8;
            float v = 0.f;
            if (col < 3)      v = dg[(pt0 + row) * 3 + col];
            else if (col < 6) v = sigmoidf(e[r]);
            hid[row * 64 + col] = (half_t)v;
            hid[row * 64 + col + 16] = (half_t)0.f;
        }
        lds_fence();

        // ---- spec MLP: [6->32] relu [32->3] sigmoid ----
        v8f p0 = {}, p1 = {};
        {
            v16h a = load_a16(hid, 64, 0);
            p0 = WMMA(a, load_b16(WP0, 32, 0, 0), p0);
            p1 = WMMA(a, load_b16(WP0, 32, 16, 0), p1);
        }
        store_h(hid, 64, 0, p0);
        store_h(hid, 64, 16, p1);
        lds_fence();
        v8f q = {};
        {
            v16h a = load_a16(hid, 64, 0);
            q = WMMA(a, load_b16(WP1, 32, 0, 0), q);
        }

        // ---- epilogue: color = clip(sigmoid(diffuse)+specular), outputs ----
        if (col < 3) {
#pragma unroll
            for (int r = 0; r < 8; ++r) {
                int ptn = pt0 + r + sel * 8;
                float spv = sigmoidf(q[r]);
                float dif = sigmoidf(e[r]);
                float cv = fminf(dif + spv, 1.f);
                out[NPTS + ptn * 3 + col] = cv;
                out[4 * NPTS + ptn * 3 + col] = spv;
            }
        }
    }
}

// ---------------- host entry ---------------------------------------------------
extern "C" void kernel_launch(void* const* d_in, const int* in_sizes, int n_in,
                              void* d_out, int out_size, void* d_ws, size_t ws_size,
                              hipStream_t stream) {
    const float* x   = (const float*)d_in[0];
    const float* d   = (const float*)d_in[1];
    const float* eD  = (const float*)d_in[2];
    const float* eC  = (const float*)d_in[3];
    const float* sw0 = (const float*)d_in[4];
    const float* sw1 = (const float*)d_in[5];
    const float* cw0 = (const float*)d_in[6];
    const float* cw1 = (const float*)d_in[7];
    const float* cw2 = (const float*)d_in[8];
    const float* pw0 = (const float*)d_in[9];
    const float* pw1 = (const float*)d_in[10];
    half_t* Wg = (half_t*)d_ws;
    float* out = (float*)d_out;

    prep_weights<<<1, 256, 0, stream>>>(sw0, sw1, cw0, cw1, cw2, pw0, pw1, Wg);
    nerf_fused<<<NPTS / PTS_PER_BLOCK, THREADS, 0, stream>>>(x, d, eD, eC, Wg, out);
}
